// Net_91225105367820
// MI455X (gfx1250) — compile-verified, matrix operand order checked by reading, not run on hardware
//
#include <hip/hip_runtime.h>
#include <hip/hip_bf16.h>

typedef __attribute__((ext_vector_type(16))) _Float16     v16h;
typedef __attribute__((ext_vector_type(8)))  _Float16     v8h;
typedef __attribute__((ext_vector_type(8)))  float        v8f;
typedef __attribute__((ext_vector_type(8)))  unsigned int v8u;

union ABFrag { v8u u; v16h h; };

#define NPTS    16384
#define NTILES  (NPTS / 16)
#define CHUNK   16            // candidate tiles staged to LDS per step
#define RSTRIDE 24            // halves per staged LDS row (48 B, conflict-free)

__device__ __forceinline__ float silu_f(float x) { return x / (1.0f + __expf(-x)); }

// ---------------------------------------------------------------------------
// Kernel A: per-point MLP 13 -> 8 -> 16 -> 15 (+ last raw feature) = x[16]
// Writes x in f32 (edge stage), f16 (WMMA Gram), and |x|^2.
// ---------------------------------------------------------------------------
__global__ void __launch_bounds__(256) point_mlp_kernel(
    const float* __restrict__ x_pfc,
    const float* __restrict__ W1, const float* __restrict__ b1,
    const float* __restrict__ W2, const float* __restrict__ b2,
    const float* __restrict__ W3, const float* __restrict__ b3,
    float* __restrict__ xf32, _Float16* __restrict__ xf16,
    float* __restrict__ sq)
{
    __shared__ float sW1[13 * 8], sb1[8], sW2[8 * 16], sb2[16], sW3[16 * 15], sb3[15];
    for (int t = threadIdx.x; t < 13 * 8; t += blockDim.x) sW1[t] = W1[t];
    for (int t = threadIdx.x; t < 8;      t += blockDim.x) sb1[t] = b1[t];
    for (int t = threadIdx.x; t < 8 * 16; t += blockDim.x) sW2[t] = W2[t];
    for (int t = threadIdx.x; t < 16;     t += blockDim.x) sb2[t] = b2[t];
    for (int t = threadIdx.x; t < 16 * 15;t += blockDim.x) sW3[t] = W3[t];
    for (int t = threadIdx.x; t < 15;     t += blockDim.x) sb3[t] = b3[t];
    __syncthreads();

    const int i = blockIdx.x * blockDim.x + threadIdx.x;
    if (i >= NPTS) return;

    float xin[13];
    #pragma unroll
    for (int k = 0; k < 13; ++k) xin[k] = x_pfc[i * 13 + k];

    float h1[8];
    #pragma unroll
    for (int j = 0; j < 8; ++j) {
        float s = sb1[j];
        #pragma unroll
        for (int k = 0; k < 13; ++k) s = fmaf(xin[k], sW1[k * 8 + j], s);
        h1[j] = silu_f(s);
    }
    float h2[16];
    #pragma unroll
    for (int m = 0; m < 16; ++m) {
        float s = sb2[m];
        #pragma unroll
        for (int j = 0; j < 8; ++j) s = fmaf(h1[j], sW2[j * 16 + m], s);
        h2[m] = silu_f(s);
    }
    float xv[16];
    #pragma unroll
    for (int o = 0; o < 15; ++o) {
        float s = sb3[o];
        #pragma unroll
        for (int m = 0; m < 16; ++m) s = fmaf(h2[m], sW3[m * 15 + o], s);
        xv[o] = s;                     // no activation on layer 3
    }
    xv[15] = xin[12];                  // concat last raw pfc feature

    float s2 = 0.0f;
    #pragma unroll
    for (int t = 0; t < 16; ++t) {
        s2 = fmaf(xv[t], xv[t], s2);
        xf32[i * 16 + t] = xv[t];
        xf16[i * 16 + t] = (_Float16)xv[t];
    }
    sq[i] = s2;
}

// ---------------------------------------------------------------------------
// Kernel B: exact K=8 KNN via WMMA Gram tiles.
//  - wave owns a 16-point tile (N dim); block = 8 waves sharing one candidate
//    stream staged through double-buffered LDS (48 B rows: 16 feature halves
//    + (f16)sq at half 16 + 0 at half 17), global loads pipelined ahead.
//  - sq_m folded into the K=16 padding slot (A[m][16]=sq_m, B[16][n]=-0.5)
//    so d2 = sq_ours - 2*C with zero per-candidate global traffic.
//  - 2-stage accumulator pipeline: wmma(jj) issues, then ds_loads(jj+1), then
//    the VALU reduction of cacc(jj-1) fills wmma(jj)'s hazard window.
//  - top-8 as order-preserving u64 keys -> native v_min_u64/v_max_u64 chains.
//  - wave-uniform __any early-skip: expected inserts per row ~ 8*ln(N) << N.
// ---------------------------------------------------------------------------
__global__ void __launch_bounds__(256) knn_kernel(
    const _Float16* __restrict__ xf16,
    const float* __restrict__ sq,
    int* __restrict__ knn)
{
    __shared__ _Float16 sx[2][CHUNK * 16 * RSTRIDE];   // 2 x 12 KB

    const int tid   = threadIdx.x;
    const int lane  = tid & 31;
    const int lp    = lane & 15;
    const int hs    = lane >> 4;          // which K-half this lane carries
    const int k0    = hs ? 8 : 0;
    const int itile = blockIdx.x * 8 + (tid >> 5);
    const int row_i = itile * 16 + lp;    // this lane's point (N dim of WMMA)

    // B fragment: our 16 points as columns; constant over the whole scan.
    v16h bfrag;
    {
        const uint4 bb = *(const uint4*)(xf16 + row_i * 16 + k0);
        const unsigned int bsc = hs ? 0u
            : (unsigned int)__builtin_bit_cast(unsigned short, (_Float16)(-0.5f));
        ABFrag f; f.u = (v8u){bb.x, bb.y, bb.z, bb.w, bsc, 0u, 0u, 0u};
        bfrag = f.h;                                   // B[16][n] = -0.5
    }
    const float sq_ours = sq[row_i];

    unsigned long long K[8];
    #pragma unroll
    for (int k = 0; k < 8; ++k) K[k] = ~0ull;
    float worstf = 3.402823e38f;

    // process one 8-row accumulator column: d2 = sq_ours - 2*C, top-8 update
    auto process = [&](const v8f& cacc, int base) {
        float dd[8];
        float dmin = 3.402823e38f;
        #pragma unroll
        for (int v = 0; v < 8; ++v) {
            dd[v] = fmaf(-2.0f, cacc[v], sq_ours);
            dmin  = fminf(dmin, dd[v]);
        }
        if (__any((int)(dmin < worstf))) {
            #pragma unroll
            for (int v = 0; v < 8; ++v) {
                unsigned int u = __float_as_uint(dd[v]);
                u ^= (u & 0x80000000u) ? 0xFFFFFFFFu : 0x80000000u;
                const unsigned long long key =
                    ((unsigned long long)u << 32) | (unsigned int)(base + v);
                const bool t0 = key < K[7];
                K[7] = t0 ? key : K[7];
                #pragma unroll
                for (int k = 7; k > 0; --k) {         // one bubble pass keeps sorted
                    const bool s = K[k] < K[k - 1];
                    const unsigned long long klo = s ? K[k] : K[k - 1];
                    const unsigned long long khi = s ? K[k - 1] : K[k];
                    K[k - 1] = klo; K[k] = khi;
                }
            }
            unsigned int wu = (unsigned int)(K[7] >> 32);   // worst back to float
            wu = (wu & 0x80000000u) ? (wu ^ 0x80000000u) : ~wu;
            worstf = __uint_as_float(wu);
        }
    };

    // ---- stage chunk 0 (thread t stages global row t of the chunk) ----
    uint4 g0, g1; float gs;
    {
        const uint4* p = (const uint4*)(xf16 + tid * 16);
        g0 = p[0]; g1 = p[1]; gs = sq[tid];
    }
    {
        _Float16* row = &sx[0][tid * RSTRIDE];
        uint4* q = (uint4*)row;
        q[0] = g0; q[1] = g1;
        *(unsigned int*)(row + 16) =
            (unsigned int)__builtin_bit_cast(unsigned short, (_Float16)gs);
    }
    __syncthreads();

    const int NCHUNKS = NTILES / CHUNK;   // 64
    for (int c = 0; c < NCHUNKS; ++c) {
        const int b = c & 1;
        // issue next chunk's global loads before consuming this one
        if (c + 1 < NCHUNKS) {
            const int r = (c + 1) * (CHUNK * 16) + tid;
            const uint4* p = (const uint4*)(xf16 + r * 16);
            g0 = p[0]; g1 = p[1]; gs = sq[r];
            if (c + 2 < NCHUNKS)
                __builtin_prefetch(xf16 + ((c + 2) * (CHUNK * 16) + tid) * 16, 0, 1);
        }

        const _Float16* xs   = sx[b];
        const int       cbase = c * CHUNK * 16 + hs * 8;

        // prologue: fragment(0), wmma(0), loads(1)
        uint4        t4 = *(const uint4*)(xs + lp * RSTRIDE + k0);
        unsigned int t1 = *(const unsigned short*)(xs + lp * RSTRIDE + 16 + hs);
        v8f cprev;
        {
            ABFrag f; f.u = (v8u){t4.x, t4.y, t4.z, t4.w, t1, 0u, 0u, 0u};
            v8f z = {};
            cprev = __builtin_amdgcn_wmma_f32_16x16x32_f16(
                        false, f.h, false, bfrag, (short)0, z, false, false);
            t4 = *(const uint4*)(xs + (16 + lp) * RSTRIDE + k0);
            t1 = *(const unsigned short*)(xs + (16 + lp) * RSTRIDE + 16 + hs);
        }

        #pragma unroll 2
        for (int jj = 1; jj < CHUNK; ++jj) {
            ABFrag f; f.u = (v8u){t4.x, t4.y, t4.z, t4.w, t1, 0u, 0u, 0u};
            v8f z = {};
            v8f ccur = __builtin_amdgcn_wmma_f32_16x16x32_f16(
                           false, f.h, false, bfrag, (short)0, z, false, false);
            if (jj + 1 < CHUNK) {
                const int nrow = (jj + 1) * 16 + lp;
                t4 = *(const uint4*)(xs + nrow * RSTRIDE + k0);
                t1 = *(const unsigned short*)(xs + nrow * RSTRIDE + 16 + hs);
            }
            process(cprev, cbase + (jj - 1) * 16);   // fills wmma(jj) hazard window
            cprev = ccur;
        }
        process(cprev, cbase + (CHUNK - 1) * 16);

        // store pre-loaded next chunk; single barrier per chunk is sufficient
        if (c + 1 < NCHUNKS) {
            _Float16* row = &sx[b ^ 1][tid * RSTRIDE];
            uint4* q = (uint4*)row;
            q[0] = g0; q[1] = g1;
            *(unsigned int*)(row + 16) =
                (unsigned int)__builtin_bit_cast(unsigned short, (_Float16)gs);
        }
        __syncthreads();
    }

    // Lanes p and p+16 hold disjoint candidate halves for the same point: merge.
    #pragma unroll
    for (int k = 0; k < 8; ++k) {
        const unsigned long long ok = __shfl(K[k], lane + 16, 32);
        if (hs == 0) {
            const bool t0 = ok < K[7];
            K[7] = t0 ? ok : K[7];
            #pragma unroll
            for (int q = 7; q > 0; --q) {
                const bool s = K[q] < K[q - 1];
                const unsigned long long klo = s ? K[q] : K[q - 1];
                const unsigned long long khi = s ? K[q - 1] : K[q];
                K[q - 1] = klo; K[q] = khi;
            }
        }
    }
    if (hs == 0) {
        #pragma unroll
        for (int k = 0; k < 8; ++k)
            knn[row_i * 8 + k] = (int)(unsigned int)K[k];
    }
}

// ---------------------------------------------------------------------------
// Kernel C: EdgeConv message MLP as WMMA. 16 edges (= 2 points x 8 nbrs) per
// WMMA; edge_in = [x_i, x_j - x_i] built straight into the A fragment (K=32
// exactly). C layout puts all 8 edges of one point in one lane -> free mean.
// ---------------------------------------------------------------------------
__global__ void __launch_bounds__(256) edge_kernel(
    const float* __restrict__ xf32,
    const int* __restrict__ knn,
    const float* __restrict__ We,
    const float* __restrict__ be,
    float* __restrict__ feats)
{
    const int lane = threadIdx.x & 31;
    const int wave = (blockIdx.x * blockDim.x + threadIdx.x) >> 5;   // 0..511
    const int lp   = lane & 15;
    const int hs   = lane >> 4;
    const int k0   = hs ? 8 : 0;

    // B fragment from We (32x16): column n=lp, K rows {k0..k0+7, k0+16..k0+23}
    v16h bw = {};
    #pragma unroll
    for (int t = 0; t < 8; ++t) {
        bw[t]     = (_Float16)We[(k0 + t) * 16 + lp];
        bw[8 + t] = (_Float16)We[(k0 + 16 + t) * 16 + lp];
    }
    const float be_n = be[lp];

    const int GROUPS_PER_WAVE = 16;   // 8192 groups / 512 waves
    for (int t = 0; t < GROUPS_PER_WAVE; ++t) {
        const int g    = wave * GROUPS_PER_WAVE + t;  // group = 2 points, 16 edges
        const int p    = g * 2 + (lp >> 3);           // edge lp's source point
        const int slot = lp & 7;
        const int jn   = knn[p * 8 + slot];

        const float4* xi4 = (const float4*)(xf32 + p  * 16 + k0);
        const float4* xj4 = (const float4*)(xf32 + jn * 16 + k0);
        const float4 i0 = xi4[0], i1 = xi4[1];
        const float4 j0 = xj4[0], j1 = xj4[1];
        float vi[8] = { i0.x, i0.y, i0.z, i0.w, i1.x, i1.y, i1.z, i1.w };
        float vj[8] = { j0.x, j0.y, j0.z, j0.w, j1.x, j1.y, j1.z, j1.w };

        v16h a = {};
        #pragma unroll
        for (int q = 0; q < 8; ++q) {
            a[q]     = (_Float16)vi[q];              // features [0..15]  = x_i
            a[8 + q] = (_Float16)(vj[q] - vi[q]);    // features [16..31] = x_j - x_i
        }

        v8f c = {};
        c = __builtin_amdgcn_wmma_f32_16x16x32_f16(
                false, a, false, bw, (short)0, c, false, false);

        float s = 0.0f;
        #pragma unroll
        for (int v = 0; v < 8; ++v) s += silu_f(c[v] + be_n);

        const int pt = g * 2 + hs;    // lanes<16 hold edges 0..7 (point g*2), else g*2+1
        feats[pt * 16 + lp] = s * 0.125f;
    }
}

// ---------------------------------------------------------------------------
// Kernel D: head MLP 16 -> 32 (SiLU) -> 16, then concat [f, x_pfc] -> 29 cols.
// ---------------------------------------------------------------------------
__global__ void __launch_bounds__(256) head_kernel(
    const float* __restrict__ feats,
    const float* __restrict__ x_pfc,
    const float* __restrict__ Wf1, const float* __restrict__ bf1,
    const float* __restrict__ Wf2, const float* __restrict__ bf2,
    float* __restrict__ out)
{
    __shared__ float sW1[16 * 32], sb1[32], sW2[32 * 16], sb2[16];
    for (int t = threadIdx.x; t < 16 * 32; t += blockDim.x) sW1[t] = Wf1[t];
    for (int t = threadIdx.x; t < 32;      t += blockDim.x) sb1[t] = bf1[t];
    for (int t = threadIdx.x; t < 32 * 16; t += blockDim.x) sW2[t] = Wf2[t];
    for (int t = threadIdx.x; t < 16;      t += blockDim.x) sb2[t] = bf2[t];
    __syncthreads();

    const int i = blockIdx.x * blockDim.x + threadIdx.x;
    if (i >= NPTS) return;

    float ft[16];
    #pragma unroll
    for (int k = 0; k < 16; ++k) ft[k] = feats[i * 16 + k];

    float h[32];
    #pragma unroll
    for (int u = 0; u < 32; ++u) {
        float s = sb1[u];
        #pragma unroll
        for (int k = 0; k < 16; ++k) s = fmaf(ft[k], sW1[k * 32 + u], s);
        h[u] = silu_f(s);
    }
    #pragma unroll
    for (int o = 0; o < 16; ++o) {
        float s = sb2[o];
        #pragma unroll
        for (int u = 0; u < 32; ++u) s = fmaf(h[u], sW2[u * 16 + o], s);
        out[i * 29 + o] = s;           // no activation on final layer
    }
    #pragma unroll
    for (int k = 0; k < 13; ++k) out[i * 29 + 16 + k] = x_pfc[i * 13 + k];
}

// ---------------------------------------------------------------------------
extern "C" void kernel_launch(void* const* d_in, const int* in_sizes, int n_in,
                              void* d_out, int out_size, void* d_ws, size_t ws_size,
                              hipStream_t stream) {
    (void)in_sizes; (void)n_in; (void)out_size; (void)ws_size;

    const float* x_pfc = (const float*)d_in[0];
    const float* W1    = (const float*)d_in[1];
    const float* b1    = (const float*)d_in[2];
    const float* W2    = (const float*)d_in[3];
    const float* b2    = (const float*)d_in[4];
    const float* W3    = (const float*)d_in[5];
    const float* b3    = (const float*)d_in[6];
    const float* We    = (const float*)d_in[7];
    const float* be    = (const float*)d_in[8];
    const float* Wf1   = (const float*)d_in[9];
    const float* bf1   = (const float*)d_in[10];
    const float* Wf2   = (const float*)d_in[11];
    const float* bf2   = (const float*)d_in[12];
    float* out = (float*)d_out;

    // Workspace layout (all 16B-aligned): total ~3.1 MB (L2-resident)
    float*     xf32  = (float*)d_ws;                      // N*16 f32
    float*     sq    = xf32 + NPTS * 16;                  // N f32
    float*     feats = sq + NPTS;                         // N*16 f32
    int*       knn   = (int*)(feats + NPTS * 16);         // N*8 i32
    _Float16*  xf16  = (_Float16*)(knn + NPTS * 8);       // N*16 f16

    dim3 blk(256);
    point_mlp_kernel<<<dim3(NPTS / 256), blk, 0, stream>>>(
        x_pfc, W1, b1, W2, b2, W3, b3, xf32, xf16, sq);

    // 1024 row tiles / 8 waves per block = 128 blocks
    knn_kernel<<<dim3(NTILES / 8), blk, 0, stream>>>(xf16, sq, knn);

    // 8192 edge groups / 16 per wave = 512 waves = 64 blocks
    edge_kernel<<<dim3(64), blk, 0, stream>>>(xf32, knn, We, be, feats);

    head_kernel<<<dim3(NPTS / 256), blk, 0, stream>>>(
        feats, x_pfc, Wf1, bf1, Wf2, bf2, out);
}
